// NGP_2619930051147
// MI455X (gfx1250) — matrix operand order repaired and checked by reading, not compiled, not run on hardware
//
#include <hip/hip_runtime.h>
#include <hip/hip_bf16.h>

typedef __attribute__((ext_vector_type(16))) _Float16 v16h;
typedef __attribute__((ext_vector_type(2)))  _Float16 v2h;
typedef __attribute__((ext_vector_type(8)))  float    v8f;

#define L_LVL 8
#define T_TBL 1024
#define HASH_P1 2654435761u
#define HASH_P2 805459861u

union FragH { v16h h; unsigned u[8]; };

// pack two f32 -> one u32 of f16 pair: single v_cvt_pk_rtz_f16_f32
__device__ __forceinline__ unsigned pack2h(float a, float b){
  auto p = __builtin_amdgcn_cvt_pkrtz(a, b);   // __fp16 ext_vector(2)
  return __builtin_bit_cast(unsigned, p);
}

// pack + packed ReLU: v_cvt_pk_rtz_f16_f32 + v_pk_max_num_f16
__device__ __forceinline__ unsigned pack2h_relu(float a, float b){
  auto p = __builtin_amdgcn_cvt_pkrtz(a, b);
  v2h q = __builtin_bit_cast(v2h, p);
  v2h z = {};
  q = __builtin_elementwise_max(q, z);
  return __builtin_bit_cast(unsigned, q);
}

// Half-wave swap (lanes 0-15 <-> 16-31) in the VALU pipe:
// v_permlanex16_b32 with identity lane selects == SWAPX16, no LDS/dscnt.
__device__ __forceinline__ unsigned swap16(unsigned v){
  return (unsigned)__builtin_amdgcn_permlanex16(
      (int)v, (int)v, 0x76543210, 0xfedcba98, true, false);
}

__device__ __forceinline__ v8f wmma_f16(v16h a, v16h b, v8f c){
  return __builtin_amdgcn_wmma_f32_16x16x32_f16(false, a, false, b, (short)0, c, false, false);
}

// B chunk (32 K-rows x 16 points) from two 16-neuron activation tiles.
// D layout: lane holds point=lane&15, neurons r+8*(lane>>4) of its tile.
// B layout: lanes 0-15 carry K=0..15 (elem j = K j), lanes 16-31 carry K=16..31.
__device__ __forceinline__ v16h make_b2(const unsigned pk0[4], const unsigned pk1[4], bool hi){
  FragH b;
#pragma unroll
  for (int i=0;i<4;i++){
    unsigned s0 = swap16(pk0[i]);
    unsigned s1 = swap16(pk1[i]);
    b.u[i]   = hi ? s1     : pk0[i];   // elems 0..7
    b.u[4+i] = hi ? pk1[i] : s0;       // elems 8..15
  }
  return b.h;
}

// K=16 activation zero-padded to K=32 (upper half-wave contributes zeros)
__device__ __forceinline__ v16h make_b1(const unsigned pk0[4], bool hi){
  FragH b;
#pragma unroll
  for (int i=0;i<4;i++){
    unsigned s0 = swap16(pk0[i]);
    b.u[i]   = hi ? 0u : pk0[i];
    b.u[4+i] = hi ? 0u : s0;
  }
  return b.h;
}

__device__ __forceinline__ void relu_pack(const v8f d, unsigned pk[4]){
#pragma unroll
  for (int i=0;i<4;i++)
    pk[i] = pack2h_relu(d[2*i], d[2*i+1]);
}

__device__ __forceinline__ void lin_pack(const v8f d, unsigned pk[4]){
#pragma unroll
  for (int i=0;i<4;i++)
    pk[i] = pack2h(d[2*i], d[2*i+1]);
}

// A fragment = W^T: M = out neurons 16*mt..+15, K = in features kbase..kbase+31.
// A k order used here (contiguous, matches B): lanes 0-15 elem 0..15 = K kbase+0..15;
// lanes 16-31 elem 0..15 = K kbase+16..31. W stored row-major [in][out].
__device__ __forceinline__ v16h load_wT(const float* __restrict__ W, int in_dim, int out_dim,
                                        int mt, int kbase, int col, int hh){
  FragH a;
  int M = mt*16 + col;
#pragma unroll
  for (int j=0;j<8;j++){
    int K0 = kbase + 16*hh + j;
    int K1 = kbase + 16*hh + 8 + j;
    float v0 = (K0 < in_dim && M < out_dim) ? W[K0*out_dim + M] : 0.f;
    float v1 = (K1 < in_dim && M < out_dim) ? W[K1*out_dim + M] : 0.f;
    a.h[j]   = (_Float16)v0;
    a.h[8+j] = (_Float16)v1;
  }
  return a.h;
}

__global__ __launch_bounds__(256) void ngp_fused(
    const float* __restrict__ X, const float* __restrict__ TBL,
    const float* __restrict__ w1, const float* __restrict__ w2,
    const float* __restrict__ r1, const float* __restrict__ r2,
    const float* __restrict__ r3, float* __restrict__ out, int N)
{
  // 8 levels x 1024 entries x float2 = 64 KB hash table in LDS
  __shared__ float2 sTbl[L_LVL * T_TBL];
  const int tid = threadIdx.x;
  for (int i = tid; i < L_LVL*T_TBL; i += 256)
    sTbl[i] = ((const float2*)TBL)[i];

  const int lane = tid & 31;
  const int col  = lane & 15;   // point within 16-point tile
  const int hh   = lane >> 4;   // half-wave id
  const bool hi  = hh != 0;

  // All weight A-fragments held in registers (20 x v16h = 160 VGPRs)
  v16h Aw1[4], Aw2[2], Ar1[4], Ar2[4][2], Ar3[2];
#pragma unroll
  for (int m=0;m<4;m++) Aw1[m] = load_wT(w1, 16, 64, m, 0, col, hh);
  Aw2[0] = load_wT(w2, 64, 16, 0,  0, col, hh);
  Aw2[1] = load_wT(w2, 64, 16, 0, 32, col, hh);
#pragma unroll
  for (int m=0;m<4;m++) Ar1[m] = load_wT(r1, 16, 64, m, 0, col, hh);
#pragma unroll
  for (int m=0;m<4;m++){
    Ar2[m][0] = load_wT(r2, 64, 64, m,  0, col, hh);
    Ar2[m][1] = load_wT(r2, 64, 64, m, 32, col, hh);
  }
  Ar3[0] = load_wT(r3, 64, 1, 0,  0, col, hh);
  Ar3[1] = load_wT(r3, 64, 1, 0, 32, col, hh);

  __syncthreads();

  const int nTiles = (N + 15) >> 4;
  const int wavesPerGrid = (gridDim.x * blockDim.x) >> 5;
  const int wv = (blockIdx.x * blockDim.x + tid) >> 5;

  // per-level resolutions: floor(2 * 5^(l/7)); half 0 -> levels 0-3, half 1 -> 4-7
  const float resLo[4] = {2.f, 2.f, 3.f, 3.f};
  const float resHi[4] = {5.f, 6.f, 7.f, 10.f};

  v8f zc = {};

  for (int t = wv; t < nTiles; t += wavesPerGrid){
    int p  = t*16 + col;
    int pc = p < N ? p : N-1;
    float xv = X[3*pc+0] + 0.5f;
    float yv = X[3*pc+1] + 0.5f;
    float zv = X[3*pc+2] + 0.5f;
    { // prefetch next tile's coordinates (global_prefetch_b8)
      long np = 3l*((long)(t + wavesPerGrid)*16 + col);
      if (np < 3l*N) __builtin_prefetch(&X[np], 0, 3);
    }

    // ---- hash-grid encode: this lane handles 4 levels of point `col` ----
    unsigned pkE[4];
#pragma unroll
    for (int l=0;l<4;l++){
      int   lvl = 4*hh + l;
      float rr  = hi ? resHi[l] : resLo[l];
      float px = xv*rr, py = yv*rr, pzc = zv*rr;
      float fx = floorf(px), fy = floorf(py), fz = floorf(pzc);
      float wx = px-fx, wy = py-fy, wz = pzc-fz;
      unsigned ix = (unsigned)(int)fx, iy = (unsigned)(int)fy, iz = (unsigned)(int)fz;
      float f0 = 0.f, f1 = 0.f;
#pragma unroll
      for (int c=0;c<8;c++){
        unsigned cx = c & 1u, cy = (c>>1)&1u, cz = (unsigned)c>>2;
        unsigned hsh = (ix+cx) ^ ((iy+cy)*HASH_P1) ^ ((iz+cz)*HASH_P2);
        float2 ft = sTbl[lvl*T_TBL + (hsh & (T_TBL-1))];
        float wgt = (cx?wx:1.f-wx) * (cy?wy:1.f-wy) * (cz?wz:1.f-wz);
        f0 = fmaf(wgt, ft.x, f0);
        f1 = fmaf(wgt, ft.y, f1);
      }
      pkE[l] = pack2h(f0, f1);
    }

    // ---- fused MLP: Z^T = W^T X^T at every layer, activations stay in regs ----
    v16h B1 = make_b1(pkE, hi);                 // h^T: 16(K) x 16(pts), pad K32
    unsigned pz4[4][4];
#pragma unroll
    for (int m=0;m<4;m++){                      // 16 -> 64, ReLU
      v8f Z = wmma_f16(Aw1[m], B1, zc);
      relu_pack(Z, pz4[m]);
    }
    v16h B2a = make_b2(pz4[0], pz4[1], hi);
    v16h B2b = make_b2(pz4[2], pz4[3], hi);

    v8f Y = wmma_f16(Aw2[0], B2a, zc);          // 64 -> 16, linear
    Y     = wmma_f16(Aw2[1], B2b, Y);
    unsigned pY[4];
    lin_pack(Y, pY);
    v16h B3 = make_b1(pY, hi);

    unsigned pg[4][4];
#pragma unroll
    for (int m=0;m<4;m++){                      // 16 -> 64, ReLU
      v8f G = wmma_f16(Ar1[m], B3, zc);
      relu_pack(G, pg[m]);
    }
    v16h B4a = make_b2(pg[0], pg[1], hi);
    v16h B4b = make_b2(pg[2], pg[3], hi);

    unsigned ph[4][4];
#pragma unroll
    for (int m=0;m<4;m++){                      // 64 -> 64, ReLU
      v8f H = wmma_f16(Ar2[m][0], B4a, zc);
      H     = wmma_f16(Ar2[m][1], B4b, H);
      relu_pack(H, ph[m]);
    }
    v16h B5a = make_b2(ph[0], ph[1], hi);
    v16h B5b = make_b2(ph[2], ph[3], hi);

    v8f O = wmma_f16(Ar3[0], B5a, zc);          // 64 -> 1 (row M=0), sigmoid
    O     = wmma_f16(Ar3[1], B5b, O);

    if (!hi && p < N){
      float o = O[0];                            // D VGPR0, lanes 0-15 = row M=0
      out[p] = 1.f / (1.f + __expf(-o));
    }
  }
}

extern "C" void kernel_launch(void* const* d_in, const int* in_sizes, int n_in,
                              void* d_out, int out_size, void* d_ws, size_t ws_size,
                              hipStream_t stream) {
  const float* X   = (const float*)d_in[0];
  const float* TBL = (const float*)d_in[1];
  const float* w1  = (const float*)d_in[2];
  const float* w2  = (const float*)d_in[3];
  const float* r1  = (const float*)d_in[4];
  const float* r2  = (const float*)d_in[5];
  const float* r3  = (const float*)d_in[6];
  float* out = (float*)d_out;

  int N = in_sizes[0] / 3;
  int nTiles = (N + 15) / 16;
  int blocks = (nTiles + 7) / 8;          // 8 waves per 256-thread block
  if (blocks > 2048) blocks = 2048;       // grid-stride: ~8 tiles/wave at N=2M,
                                          // amortizes the weight-fragment prologue
  if (blocks < 1) blocks = 1;

  hipLaunchKernelGGL(ngp_fused, dim3(blocks), dim3(256), 0, stream,
                     X, TBL, w1, w2, r1, r2, r3, out, N);
}
